// DeformNet_EPF1_60868276519448
// MI455X (gfx1250) — matrix operand order repaired and checked by reading, not compiled, not run on hardware
//
#include <hip/hip_runtime.h>
#include <hip/hip_bf16.h>
#include <stdint.h>

// ---------------------------------------------------------------------------
// Types for CDNA5 WMMA (wave32): V_WMMA_F32_16X16X32_BF16
// A: 16(M)x32(K) bf16 -> 16 bf16/lane ; B: 32(K)x16(N) bf16 -> 16 bf16/lane
// C/D: 16x16 f32 -> 8 f32/lane
// ---------------------------------------------------------------------------
typedef __attribute__((ext_vector_type(16))) __bf16 v16bf;
typedef __attribute__((ext_vector_type(8)))  float  v8f;

union FragBF {
    uint4 u[2];   // 2 x 16B = 32B
    v16bf v;
};

static __device__ __forceinline__ uint16_t f32_to_bf16(float f) {
    uint32_t u = __float_as_uint(f);
    uint32_t r = (u + 0x7FFFu + ((u >> 16) & 1u)) >> 16;   // RNE
    return (uint16_t)r;
}

// ---------------------------------------------------------------------------
// Blocked GEMM: C[M,N] = A[M,Kpad](bf16) x W[Npad64,Kpad](bf16)^T
// fused affine + relu epilogue:  out[m,n] = relu(acc * scale[n] + shift[n])
//
// Each wave computes a 16(M) x 64(N) tile: one A fragment feeds 4 WMMAs.
// The k-loop is manually 2x-unrolled into a ping-pong so the two fragment
// register sets alternate compute/load roles with NO register copies
// (the compiler lowered `a = an; ...` as 40 v_dual_mov per iteration).
//
// Per-lane fragment addressing (CDNA5 ISA 16-bit layouts):
//   A: lane(0..15)->M=lane, half=0 ; lane(16..31)->M=lane-16, half=1
//      e=0..7  : K = k0 + 8*half + e        (contiguous 8)
//      e=8..15 : K = k0 + 16 + 8*half + (e-8)
//   B: lane -> N=lane&15, khalf=lane>>4
//      e=0..7  : K = k0 + 8*khalf + e       (same as A)
//      e=8..15 : K = k0 + 16 + 8*khalf + (e-8)
// ---------------------------------------------------------------------------
__global__ void gemm_bf16_wmma4(const uint16_t* __restrict__ A,
                                const uint16_t* __restrict__ Wt,
                                float* __restrict__ C,
                                const float* __restrict__ scale,
                                const float* __restrict__ shift,
                                int M, int N, int Npad64, int Kpad, int relu)
{
    const int lane = threadIdx.x & 31;
    const int wid  = blockIdx.x * (blockDim.x >> 5) + (threadIdx.x >> 5);
    const int ng   = Npad64 >> 6;            // 64-wide N groups
    const int total = (M >> 4) * ng;
    if (wid >= total) return;

    const int mt = wid / ng;
    const int g  = wid - mt * ng;
    const int half = lane >> 4;
    const int l15  = lane & 15;

    const uint16_t* arow = A  + (size_t)(mt * 16 + l15) * (size_t)Kpad;
    const uint16_t* brow = Wt + (size_t)(g * 64 + l15) * (size_t)Kpad;
    const size_t bstep = (size_t)16 * (size_t)Kpad;

    v8f acc0 = {0.f,0.f,0.f,0.f,0.f,0.f,0.f,0.f};
    v8f acc1 = acc0, acc2 = acc0, acc3 = acc0;

    FragBF a, b0, b1, b2, b3;        // buffer set P
    FragBF an, c0, c1, c2, c3;       // buffer set Q

#define LOAD_A(f, k) do {                                                     \
        (f).u[0] = *reinterpret_cast<const uint4*>(arow + (k) + 8 * half);    \
        (f).u[1] = *reinterpret_cast<const uint4*>(arow + (k) + 16 + 8*half); \
    } while (0)
#define LOAD_B(f, j, k) do {                                                  \
        const uint16_t* p = brow + (size_t)(j) * bstep + (k) + 8 * half;      \
        (f).u[0] = *reinterpret_cast<const uint4*>(p);                        \
        (f).u[1] = *reinterpret_cast<const uint4*>(p + 16);                   \
    } while (0)
#define WMMA4(fa, f0, f1, f2, f3) do {                                        \
        acc0 = __builtin_amdgcn_wmma_f32_16x16x32_bf16(false, (fa).v, false, (f0).v, (short)0, acc0, false, false); \
        acc1 = __builtin_amdgcn_wmma_f32_16x16x32_bf16(false, (fa).v, false, (f1).v, (short)0, acc1, false, false); \
        acc2 = __builtin_amdgcn_wmma_f32_16x16x32_bf16(false, (fa).v, false, (f2).v, (short)0, acc2, false, false); \
        acc3 = __builtin_amdgcn_wmma_f32_16x16x32_bf16(false, (fa).v, false, (f3).v, (short)0, acc3, false, false); \
    } while (0)

    LOAD_A(a, 0);
    LOAD_B(b0, 0, 0); LOAD_B(b1, 1, 0); LOAD_B(b2, 2, 0); LOAD_B(b3, 3, 0);

    // Ping-pong over k: set P computes while set Q loads, then roles swap.
    int k0 = 0;
    for (;;) {
        int kn = k0 + 32;
        if (kn < Kpad) {                         // uniform across wave
            __builtin_prefetch(arow + kn + 32, 0, 1);   // global_prefetch_b8
            LOAD_A(an, kn);
            LOAD_B(c0, 0, kn); LOAD_B(c1, 1, kn);
            LOAD_B(c2, 2, kn); LOAD_B(c3, 3, kn);
        }
        WMMA4(a, b0, b1, b2, b3);                // compute on set P
        if (kn >= Kpad) break;
        k0 = kn;

        kn = k0 + 32;
        if (kn < Kpad) {
            __builtin_prefetch(arow + kn + 32, 0, 1);
            LOAD_A(a, kn);
            LOAD_B(b0, 0, kn); LOAD_B(b1, 1, kn);
            LOAD_B(b2, 2, kn); LOAD_B(b3, 3, kn);
        }
        WMMA4(an, c0, c1, c2, c3);               // compute on set Q
        if (kn >= Kpad) break;
        k0 = kn;
    }
#undef LOAD_A
#undef LOAD_B
#undef WMMA4

    const v8f accs[4] = {acc0, acc1, acc2, acc3};
    #pragma unroll
    for (int j = 0; j < 4; ++j) {
        const int n = (g * 4 + j) * 16 + l15;
        if (n < N) {
            const float s  = scale[n];
            const float sh = shift[n];
            #pragma unroll
            for (int r = 0; r < 8; ++r) {
                const int m = mt * 16 + r + 8 * half;  // C/D layout: VGPR r -> M=r(+8)
                float v = accs[j][r] * s + sh;
                if (relu) v = fmaxf(v, 0.f);
                C[(size_t)m * (size_t)N + n] = v;
            }
        }
    }
}

// ---------------------------------------------------------------------------
// im2col for 3x3 pad-1 conv, 2D grid: blockIdx.y = m (spatial position),
// threads over k. No div/mod by Kpad needed.
// src addressed with arbitrary strides (works for NCHW input and NHWC acts).
// A[m, k]: m=(b,h,w), k = c*9 + (ki*3+kj); zero-pad k in [K, Kpad).
// ---------------------------------------------------------------------------
__global__ void im2col3x3(const float* __restrict__ src, uint16_t* __restrict__ A,
                          int H, int W,
                          long sb, long sc, long sh, long sw,
                          int K, int Kpad)
{
    const int  k = blockIdx.x * blockDim.x + threadIdx.x;
    const long m = blockIdx.y;
    if (k >= Kpad) return;
    uint16_t out = 0;
    if (k < K) {
        const int c  = k / 9;
        const int n  = k - c * 9;
        const int di = n / 3 - 1;
        const int dj = n % 3 - 1;
        const int w  = (int)(m % W);
        const int h  = (int)((m / W) % H);
        const int b  = (int)(m / ((long)W * H));
        const int hh = h + di, ww = w + dj;
        float v = 0.f;
        if (hh >= 0 && hh < H && ww >= 0 && ww < W)
            v = src[(long)b * sb + (long)c * sc + (long)hh * sh + (long)ww * sw];
        out = f32_to_bf16(v);
    }
    A[m * Kpad + k] = out;
}

// ---------------------------------------------------------------------------
// Deformable im2col (2D grid like above): bilinear sampling on virtually
// pad-1 padded feature map. off is [b,h,w,18] NHWC f32 (0..8 row, 9..17 col).
// ---------------------------------------------------------------------------
__global__ void deform_im2col(const float* __restrict__ src,
                              const float* __restrict__ off,
                              uint16_t* __restrict__ A,
                              int H, int W,
                              long sb, long sc, long sh, long sw,
                              int K, int Kpad)
{
    const int  k = blockIdx.x * blockDim.x + threadIdx.x;
    const long m = blockIdx.y;
    if (k >= Kpad) return;
    const int Hp = H + 2, Wp = W + 2;
    uint16_t out = 0;
    if (k < K) {
        const int c = k / 9;
        const int n = k - c * 9;
        const int w = (int)(m % W);
        const int h = (int)((m / W) % H);
        const int b = (int)(m / ((long)W * H));

        const float ox = off[m * 18 + n];
        const float oy = off[m * 18 + 9 + n];
        float px = (float)(h + 1) + (float)(n / 3 - 1) + ox;
        float py = (float)(w + 1) + (float)(n % 3 - 1) + oy;
        px = fminf(fmaxf(px, 0.f), (float)(Hp - 1));
        py = fminf(fmaxf(py, 0.f), (float)(Wp - 1));
        const float x0 = floorf(px);
        const float y0 = floorf(py);
        const float x1 = fminf(x0 + 1.f, (float)(Hp - 1));
        const float y1 = fminf(y0 + 1.f, (float)(Wp - 1));
        const float g_lt = (1.f + x0 - px) * (1.f + y0 - py);
        const float g_rb = (1.f - x1 + px) * (1.f - y1 + py);
        const float g_lb = (1.f + x0 - px) * (1.f - y1 + py);
        const float g_rt = (1.f - x1 + px) * (1.f + y0 - py);
        const int ix0 = (int)x0, iy0 = (int)y0, ix1 = (int)x1, iy1 = (int)y1;

        const long base = (long)b * sb + (long)c * sc;
        auto fetch = [&](int ix, int iy) -> float {
            ix -= 1; iy -= 1;                       // padded -> unpadded
            if (ix < 0 || ix >= H || iy < 0 || iy >= W) return 0.f;
            return src[base + (long)ix * sh + (long)iy * sw];
        };
        const float v = g_lt * fetch(ix0, iy0) + g_rb * fetch(ix1, iy1)
                      + g_lb * fetch(ix0, iy1) + g_rt * fetch(ix1, iy0);
        out = f32_to_bf16(v);
    }
    A[m * Kpad + k] = out;
}

// ---------------------------------------------------------------------------
// 2x2 pool on NHWC f32 (stride 2), max or avg.
// ---------------------------------------------------------------------------
__global__ void pool2x2(const float* __restrict__ src, float* __restrict__ dst,
                        int Bn, int H, int W, int Cc, int is_max)
{
    const int Ho = H >> 1, Wo = W >> 1;
    const long total = (long)Bn * Ho * Wo * Cc;
    for (long i = (long)blockIdx.x * blockDim.x + threadIdx.x; i < total;
         i += (long)gridDim.x * blockDim.x) {
        const int c = (int)(i % Cc);
        long t = i / Cc;
        const int w = (int)(t % Wo); t /= Wo;
        const int h = (int)(t % Ho);
        const int b = (int)(t / Ho);
        const long r0 = (((long)b * H + 2 * h) * W + 2 * w) * Cc + c;
        const long r1 = r0 + (long)W * Cc;
        const float a0 = src[r0], a1 = src[r0 + Cc], a2 = src[r1], a3 = src[r1 + Cc];
        dst[i] = is_max ? fmaxf(fmaxf(a0, a1), fmaxf(a2, a3))
                        : (a0 + a1 + a2 + a3) * 0.25f;
    }
}

// ---------------------------------------------------------------------------
// Pack f32 weight (N rows of length K, row-major) into bf16 [Npad64, Kpad].
// OIHW 3x3 weights flatten to rows of length IC*9 with k = ic*9 + tap, which
// matches the im2col column ordering exactly. Also used for FC weights.
// ---------------------------------------------------------------------------
__global__ void pack_weight(const float* __restrict__ w, uint16_t* __restrict__ Wb,
                            int N, int K, int Npad, int Kpad)
{
    const long total = (long)Npad * Kpad;
    for (long i = (long)blockIdx.x * blockDim.x + threadIdx.x; i < total;
         i += (long)gridDim.x * blockDim.x) {
        const int n = (int)(i / Kpad);
        const int k = (int)(i - (long)n * Kpad);
        Wb[i] = (n < N && k < K) ? f32_to_bf16(w[(long)n * K + k]) : (uint16_t)0;
    }
}

// Pack f32 activation [M,N] -> bf16 [M,Kpad], zero-padded columns.
__global__ void pack_act(const float* __restrict__ src, uint16_t* __restrict__ dst,
                         int M, int N, int Kpad)
{
    const long total = (long)M * Kpad;
    for (long i = (long)blockIdx.x * blockDim.x + threadIdx.x; i < total;
         i += (long)gridDim.x * blockDim.x) {
        const int m = (int)(i / Kpad);
        const int k = (int)(i - (long)m * Kpad);
        dst[i] = (k < N) ? f32_to_bf16(src[(long)m * N + k]) : (uint16_t)0;
    }
}

// NHWC [B,H,W,C] f32 -> bf16 A [B, C*H*W] in NCHW flatten order (for FC).
__global__ void flatten_nchw(const float* __restrict__ src, uint16_t* __restrict__ dst,
                             int Bn, int Cc, int H, int W)
{
    const int chw = Cc * H * W;
    const long total = (long)Bn * chw;
    for (long i = (long)blockIdx.x * blockDim.x + threadIdx.x; i < total;
         i += (long)gridDim.x * blockDim.x) {
        const int kk = (int)(i % chw);
        const int b  = (int)(i / chw);
        const int c  = kk / (H * W);
        const int r  = kk - c * (H * W);
        const int h  = r / W;
        const int w  = r - h * W;
        dst[i] = f32_to_bf16(src[(((long)b * H + h) * W + w) * Cc + c]);
    }
}

// BN folding: scale = g*rsqrt(v+eps); shift = b_bn + (conv_bias - mean)*scale
__global__ void make_affine_bn(const float* __restrict__ g, const float* __restrict__ bb,
                               const float* __restrict__ mn, const float* __restrict__ vr,
                               const float* __restrict__ convb,
                               float* __restrict__ scale, float* __restrict__ shift,
                               int n, int has_convb)
{
    const int i = blockIdx.x * blockDim.x + threadIdx.x;
    if (i < n) {
        const float s = g[i] * rsqrtf(vr[i] + 1e-5f);
        scale[i] = s;
        shift[i] = bb[i] + ((has_convb ? convb[i] : 0.f) - mn[i]) * s;
    }
}

// Plain bias epilogue: scale=1, shift=bias.
__global__ void make_affine_bias(const float* __restrict__ b,
                                 float* __restrict__ scale, float* __restrict__ shift, int n)
{
    const int i = blockIdx.x * blockDim.x + threadIdx.x;
    if (i < n) { scale[i] = 1.f; shift[i] = b[i]; }
}

// ---------------------------------------------------------------------------
// Host-side orchestration
// ---------------------------------------------------------------------------
extern "C" void kernel_launch(void* const* d_in, const int* in_sizes, int n_in,
                              void* d_out, int out_size, void* d_ws, size_t ws_size,
                              hipStream_t stream)
{
    const float* x     = (const float*)d_in[0];
    const float* d1_pw = (const float*)d_in[1];
    const float* d1_pb = (const float*)d_in[2];
    const float* d1_w  = (const float*)d_in[3];
    const float* bn1_g = (const float*)d_in[4];
    const float* bn1_b = (const float*)d_in[5];
    const float* bn1_m = (const float*)d_in[6];
    const float* bn1_v = (const float*)d_in[7];
    const float* c2_w  = (const float*)d_in[8];
    const float* c2_b  = (const float*)d_in[9];
    const float* bn2_g = (const float*)d_in[10];
    const float* bn2_b = (const float*)d_in[11];
    const float* bn2_m = (const float*)d_in[12];
    const float* bn2_v = (const float*)d_in[13];
    const float* d3_pw = (const float*)d_in[14];
    const float* d3_pb = (const float*)d_in[15];
    const float* d3_w  = (const float*)d_in[16];
    const float* bn3_g = (const float*)d_in[17];
    const float* bn3_b = (const float*)d_in[18];
    const float* bn3_m = (const float*)d_in[19];
    const float* bn3_v = (const float*)d_in[20];
    const float* c4_w  = (const float*)d_in[21];
    const float* c4_b  = (const float*)d_in[22];
    const float* bn4_g = (const float*)d_in[23];
    const float* bn4_b = (const float*)d_in[24];
    const float* bn4_m = (const float*)d_in[25];
    const float* bn4_v = (const float*)d_in[26];
    const float* c5_w  = (const float*)d_in[27];
    const float* c5_b  = (const float*)d_in[28];
    const float* bn5_g = (const float*)d_in[29];
    const float* bn5_b = (const float*)d_in[30];
    const float* bn5_m = (const float*)d_in[31];
    const float* bn5_v = (const float*)d_in[32];
    const float* c6_w  = (const float*)d_in[33];
    const float* c6_b  = (const float*)d_in[34];
    const float* bn6_g = (const float*)d_in[35];
    const float* bn6_b = (const float*)d_in[36];
    const float* bn6_m = (const float*)d_in[37];
    const float* bn6_v = (const float*)d_in[38];
    const float* fc_w  = (const float*)d_in[39];
    const float* fc_b  = (const float*)d_in[40];
    const float* fc1_w = (const float*)d_in[41];
    const float* fc1_b = (const float*)d_in[42];

    const int B = 128;
    char* ws = (char*)d_ws;

    // Workspace layout (bytes, 256-aligned by construction)
    const size_t SZ_A   = (size_t)32768 * 1824 * 2;   // 119,537,664 (max im2col, bf16)
    const size_t SZ_W   = 512 * 1024;                 // packed weights (bf16), max 128*1824*2
    const size_t SZ_AFF = 4096;                       // scale[256] + shift[256]
    const size_t SZ_OFF = (size_t)32768 * 18 * 4;     // offsets NHWC f32
    const size_t SZ_H1  = (size_t)32768 * 96 * 4;
    const size_t SZ_H2  = SZ_H1;
    const size_t SZ_P2  = (size_t)8192 * 96 * 4;
    const size_t SZ_H3  = (size_t)8192 * 108 * 4;
    const size_t SZ_H4  = SZ_H3;
    const size_t SZ_P4  = (size_t)2048 * 108 * 4;
    const size_t SZ_H5  = (size_t)2048 * 128 * 4;
    const size_t SZ_H6  = SZ_H5;
    const size_t SZ_P6  = (size_t)512 * 128 * 4;

    size_t off = 0;
    uint16_t* A    = (uint16_t*)(ws + off); off += SZ_A;
    uint16_t* Wb   = (uint16_t*)(ws + off); off += SZ_W;
    float* scale   = (float*)(ws + off);
    float* shift   = scale + 256;            off += SZ_AFF;
    float* offb    = (float*)(ws + off);     off += SZ_OFF;
    float* h1      = (float*)(ws + off);     off += SZ_H1;
    float* h2      = (float*)(ws + off);     off += SZ_H2;
    float* p2      = (float*)(ws + off);     off += SZ_P2;
    float* h3      = (float*)(ws + off);     off += SZ_H3;
    float* h4      = (float*)(ws + off);     off += SZ_H4;
    float* p4      = (float*)(ws + off);     off += SZ_P4;
    float* h5      = (float*)(ws + off);     off += SZ_H5;
    float* h6      = (float*)(ws + off);     off += SZ_H6;
    float* p6      = (float*)(ws + off);     off += SZ_P6;
    float* fco     = (float*)(ws + off);     off += (size_t)128 * 200 * 4;
    (void)ws_size; (void)in_sizes; (void)n_in; (void)out_size;

    auto blocks_for = [](long total) -> int {
        long b = (total + 255) / 256;
        if (b > (1L << 20)) b = (1L << 20);
        return (int)b;
    };
    // GEMM launcher: Npad64 must be a multiple of 64 (weight rows zero-padded)
    auto gemm = [&](const uint16_t* Am, const uint16_t* Wm, float* Cm,
                    int M, int N, int Npad64, int Kpad, int relu) {
        const int total = (M / 16) * (Npad64 / 64);
        const int blocks = (total + 7) / 8;       // 8 waves (16x64 tiles) / block
        gemm_bf16_wmma4<<<blocks, 256, 0, stream>>>(Am, Wm, Cm, scale, shift,
                                                    M, N, Npad64, Kpad, relu);
    };
    auto i2c_grid = [](int M, int Kpad) { return dim3((Kpad + 255) / 256, M); };

    // ---- Layer 1: deform block (200 -> 96) @16x16, M = 32768 -------------
    // 1a. offset conv 200->18 (bias, no bn/relu)
    pack_weight<<<blocks_for(64L * 1824), 256, 0, stream>>>(d1_pw, Wb, 18, 1800, 64, 1824);
    make_affine_bias<<<1, 64, 0, stream>>>(d1_pb, scale, shift, 18);
    im2col3x3<<<i2c_grid(32768, 1824), 256, 0, stream>>>(
        x, A, 16, 16, 200L * 256, 256, 16, 1, 1800, 1824);
    gemm(A, Wb, offb, 32768, 18, 64, 1824, 0);
    // 1b. deform gather + 200->96 GEMM + bn1 + relu
    pack_weight<<<blocks_for(128L * 1824), 256, 0, stream>>>(d1_w, Wb, 96, 1800, 128, 1824);
    make_affine_bn<<<1, 128, 0, stream>>>(bn1_g, bn1_b, bn1_m, bn1_v, nullptr, scale, shift, 96, 0);
    deform_im2col<<<i2c_grid(32768, 1824), 256, 0, stream>>>(
        x, offb, A, 16, 16, 200L * 256, 256, 16, 1, 1800, 1824);
    gemm(A, Wb, h1, 32768, 96, 128, 1824, 1);

    // ---- Layer 2: conv 96->96 + bn2 + relu + maxpool ---------------------
    pack_weight<<<blocks_for(128L * 896), 256, 0, stream>>>(c2_w, Wb, 96, 864, 128, 896);
    make_affine_bn<<<1, 128, 0, stream>>>(bn2_g, bn2_b, bn2_m, bn2_v, c2_b, scale, shift, 96, 1);
    im2col3x3<<<i2c_grid(32768, 896), 256, 0, stream>>>(
        h1, A, 16, 16, 16L * 16 * 96, 1, 16L * 96, 96, 864, 896);
    gemm(A, Wb, h2, 32768, 96, 128, 896, 1);
    pool2x2<<<blocks_for(8192L * 96), 256, 0, stream>>>(h2, p2, B, 16, 16, 96, 1);

    // ---- Layer 3: deform block (96 -> 108) @8x8, M = 8192 ----------------
    pack_weight<<<blocks_for(64L * 896), 256, 0, stream>>>(d3_pw, Wb, 18, 864, 64, 896);
    make_affine_bias<<<1, 64, 0, stream>>>(d3_pb, scale, shift, 18);
    im2col3x3<<<i2c_grid(8192, 896), 256, 0, stream>>>(
        p2, A, 8, 8, 8L * 8 * 96, 1, 8L * 96, 96, 864, 896);
    gemm(A, Wb, offb, 8192, 18, 64, 896, 0);

    pack_weight<<<blocks_for(128L * 896), 256, 0, stream>>>(d3_w, Wb, 108, 864, 128, 896);
    make_affine_bn<<<1, 128, 0, stream>>>(bn3_g, bn3_b, bn3_m, bn3_v, nullptr, scale, shift, 108, 0);
    deform_im2col<<<i2c_grid(8192, 896), 256, 0, stream>>>(
        p2, offb, A, 8, 8, 8L * 8 * 96, 1, 8L * 96, 96, 864, 896);
    gemm(A, Wb, h3, 8192, 108, 128, 896, 1);

    // ---- Layer 4: conv 108->108 + bn4 + relu + maxpool -------------------
    pack_weight<<<blocks_for(128L * 992), 256, 0, stream>>>(c4_w, Wb, 108, 972, 128, 992);
    make_affine_bn<<<1, 128, 0, stream>>>(bn4_g, bn4_b, bn4_m, bn4_v, c4_b, scale, shift, 108, 1);
    im2col3x3<<<i2c_grid(8192, 992), 256, 0, stream>>>(
        h3, A, 8, 8, 8L * 8 * 108, 1, 8L * 108, 108, 972, 992);
    gemm(A, Wb, h4, 8192, 108, 128, 992, 1);
    pool2x2<<<blocks_for(2048L * 108), 256, 0, stream>>>(h4, p4, B, 8, 8, 108, 1);

    // ---- Layer 5: conv 108->128 + bn5 + relu, M = 2048 -------------------
    pack_weight<<<blocks_for(128L * 992), 256, 0, stream>>>(c5_w, Wb, 128, 972, 128, 992);
    make_affine_bn<<<1, 128, 0, stream>>>(bn5_g, bn5_b, bn5_m, bn5_v, c5_b, scale, shift, 128, 1);
    im2col3x3<<<i2c_grid(2048, 992), 256, 0, stream>>>(
        p4, A, 4, 4, 4L * 4 * 108, 1, 4L * 108, 108, 972, 992);
    gemm(A, Wb, h5, 2048, 128, 128, 992, 1);

    // ---- Layer 6: conv 128->128 + bn6 + relu + avgpool -------------------
    pack_weight<<<blocks_for(128L * 1152), 256, 0, stream>>>(c6_w, Wb, 128, 1152, 128, 1152);
    make_affine_bn<<<1, 128, 0, stream>>>(bn6_g, bn6_b, bn6_m, bn6_v, c6_b, scale, shift, 128, 1);
    im2col3x3<<<i2c_grid(2048, 1152), 256, 0, stream>>>(
        h5, A, 4, 4, 4L * 4 * 128, 1, 4L * 128, 128, 1152, 1152);
    gemm(A, Wb, h6, 2048, 128, 128, 1152, 1);
    pool2x2<<<blocks_for(512L * 128), 256, 0, stream>>>(h6, p6, B, 4, 4, 128, 0);

    // ---- FC: flatten (NCHW order) -> [128,512] x fc_w[200,512]^T + fc_b --
    flatten_nchw<<<blocks_for(128L * 512), 256, 0, stream>>>(p6, A, B, 128, 2, 2);
    pack_weight<<<blocks_for(256L * 512), 256, 0, stream>>>(fc_w, Wb, 200, 512, 256, 512);
    make_affine_bias<<<1, 256, 0, stream>>>(fc_b, scale, shift, 200);
    gemm(A, Wb, fco, 128, 200, 256, 512, 0);

    // ---- FC1: [128,200] x fc1_w[16,200]^T + fc1_b -> d_out ---------------
    pack_act<<<blocks_for(128L * 224), 256, 0, stream>>>(fco, A, 128, 200, 224);
    pack_weight<<<blocks_for(64L * 224), 256, 0, stream>>>(fc1_w, Wb, 16, 200, 64, 224);
    make_affine_bias<<<1, 64, 0, stream>>>(fc1_b, scale, shift, 16);
    gemm(A, Wb, (float*)d_out, 128, 16, 64, 224, 0);
}
